// SDPAttention_25744033972760
// MI455X (gfx1250) — matrix operand order repaired and checked by reading, not compiled, not run on hardware
//
#include <hip/hip_runtime.h>

typedef __attribute__((ext_vector_type(16))) __bf16 v16bf;
typedef __attribute__((ext_vector_type(8)))  __bf16 v8bf;
typedef __attribute__((ext_vector_type(4)))  __bf16 v4bf;
typedef __attribute__((ext_vector_type(8)))  float  v8f;
typedef __attribute__((ext_vector_type(4)))  float  f32x4;

#define SEQ 8192
#define DM  128
#define BC  64                        // keys per iteration
#define BR  16                        // query rows per wave
#define WAVES 8                       // two key-split groups of 4 waves
#define THREADS (WAVES * 32)          // 256
#define GTHREADS 128                  // threads per key-split group
#define ROWS_PER_BLOCK 64
#define HSEQ (SEQ / 2)                // keys per group

// Padded LDS strides (bf16 elements) -> conflict-free ds_load windows.
#define KSTR 136   // 272 B = 68 dw; 68 % 64 = 4
#define VSTR 72    // 144 B = 36 dw; gcd(36,64) = 4
#define PSTR 72

#define KV_BYTES  (BC * KSTR * 2 + DM * VSTR * 2)   // 35840 per group
#define V_OFF     (BC * KSTR * 2)                   // 17408
#define P_OFF     (2 * KV_BYTES)                    // 71680
#define P_BYTES   (BR * PSTR * 2)                   // 2304
#define SMEM_BYTES (P_OFF + WAVES * P_BYTES)        // 90112

// 1/sqrt(8192) * log2(e): softmax computed in exp2 domain.
#define SCALE_LOG2E 0.015939679f

static __device__ __forceinline__ __bf16 f2bf(float f) { return (__bf16)f; }

template<int CTRL>
static __device__ __forceinline__ float dpp_movf(float x) {
    return __builtin_bit_cast(float,
        __builtin_amdgcn_update_dpp(0, __builtin_bit_cast(int, x),
                                    CTRL, 0xf, 0xf, true));
}
// Reductions across each 16-lane half of the wave (pure VALU DPP, no LDS).
static __device__ __forceinline__ float red_max16(float x) {
    x = fmaxf(x, dpp_movf<0xB1>(x));
    x = fmaxf(x, dpp_movf<0x4E>(x));
    x = fmaxf(x, dpp_movf<0x141>(x));
    x = fmaxf(x, dpp_movf<0x140>(x));
    return x;
}
static __device__ __forceinline__ float red_sum16(float x) {
    x += dpp_movf<0xB1>(x);
    x += dpp_movf<0x4E>(x);
    x += dpp_movf<0x141>(x);
    x += dpp_movf<0x140>(x);
    return x;
}

static __device__ __forceinline__ v16bf ld_frag(const __bf16* p) {
    v8bf lo = *(const v8bf*)p;
    v8bf hi = *(const v8bf*)(p + 16);
    return __builtin_shufflevector(lo, hi, 0,1,2,3,4,5,6,7,8,9,10,11,12,13,14,15);
}

__global__ __launch_bounds__(THREADS)
void SDPAttention_fa_kernel(const float* __restrict__ Q,
                            const float* __restrict__ K,
                            const float* __restrict__ V,
                            float* __restrict__ O)
{
    __shared__ alignas(128) unsigned char smem[SMEM_BYTES];

    const int tid  = threadIdx.x;
    const int lane = tid & 31;
    const int wave = tid >> 5;
    const int grp  = wave >> 2;       // key-split group: 0 or 1
    const int gtid = tid & (GTHREADS - 1);
    const int half = lane >> 4;
    const int l16  = lane & 15;

    __bf16* k_lds  = (__bf16*)(smem + grp * KV_BYTES);
    __bf16* vT_lds = (__bf16*)(smem + grp * KV_BYTES + V_OFF);
    __bf16* p_me   = (__bf16*)(smem + P_OFF + wave * P_BYTES);

    const int q0   = blockIdx.x * ROWS_PER_BLOCK + (wave & 3) * BR;
    const int key0 = grp * HSEQ;

    // ---- Q fragments (A layout), 4 chunks of K=32 along d ----
    v16bf qf[4];
    {
        const float* qrow = Q + (size_t)(q0 + l16) * DM;
        #pragma unroll
        for (int c = 0; c < 4; ++c) {
            const float* pa = qrow + c * 32 + half * 8;
            const float* pb = qrow + c * 32 + 16 + half * 8;
            f32x4 a0 = *(const f32x4*)(pa);
            f32x4 a1 = *(const f32x4*)(pa + 4);
            f32x4 b0 = *(const f32x4*)(pb);
            f32x4 b1 = *(const f32x4*)(pb + 4);
            #pragma unroll
            for (int j = 0; j < 4; ++j) {
                qf[c][j]      = f2bf(a0[j]);
                qf[c][4 + j]  = f2bf(a1[j]);
                qf[c][8 + j]  = f2bf(b0[j]);
                qf[c][12 + j] = f2bf(b1[j]);
            }
        }
    }

    v8f o[8] = {};
    float m[8], l[8];
    #pragma unroll
    for (int r = 0; r < 8; ++r) { m[r] = -1e30f; l[r] = 0.0f; }

    for (int kb = 0; kb < HSEQ; kb += BC) {
        const int kglob = key0 + kb;
        if (kb + BC < HSEQ) {
            __builtin_prefetch(K + (size_t)(kglob + BC) * DM, 0, 3);
            __builtin_prefetch(V + (size_t)(kglob + BC) * DM, 0, 3);
        }

        // ---- Stage K (row-major) + V (transposed) as bf16 (per group) ----
        {
            f32x4 kreg[8], vreg[8];
            #pragma unroll
            for (int i = 0; i < 8; ++i) {
                int idx = gtid + i * GTHREADS;
                int row = idx >> 5, col = (idx & 31) << 2;
                kreg[i] = *(const f32x4*)(K + (size_t)(kglob + row) * DM + col);
                vreg[i] = *(const f32x4*)(V + (size_t)(kglob + row) * DM + col);
            }
            __syncthreads();
            #pragma unroll
            for (int i = 0; i < 8; ++i) {
                int idx = gtid + i * GTHREADS;
                int row = idx >> 5, col = (idx & 31) << 2;
                v4bf k4;
                k4[0] = f2bf(kreg[i][0]); k4[1] = f2bf(kreg[i][1]);
                k4[2] = f2bf(kreg[i][2]); k4[3] = f2bf(kreg[i][3]);
                *(v4bf*)&k_lds[row * KSTR + col] = k4;
                #pragma unroll
                for (int c = 0; c < 4; ++c)
                    vT_lds[(col + c) * VSTR + row] = f2bf(vreg[i][c]);
            }
            #pragma unroll
            for (int i = 0; i < 8; ++i) {
                int idx = 1024 + gtid + i * GTHREADS;
                int row = idx >> 5, col = (idx & 31) << 2;
                kreg[i] = *(const f32x4*)(K + (size_t)(kglob + row) * DM + col);
                vreg[i] = *(const f32x4*)(V + (size_t)(kglob + row) * DM + col);
            }
            #pragma unroll
            for (int i = 0; i < 8; ++i) {
                int idx = 1024 + gtid + i * GTHREADS;
                int row = idx >> 5, col = (idx & 31) << 2;
                v4bf k4;
                k4[0] = f2bf(kreg[i][0]); k4[1] = f2bf(kreg[i][1]);
                k4[2] = f2bf(kreg[i][2]); k4[3] = f2bf(kreg[i][3]);
                *(v4bf*)&k_lds[row * KSTR + col] = k4;
                #pragma unroll
                for (int c = 0; c < 4; ++c)
                    vT_lds[(col + c) * VSTR + row] = f2bf(vreg[i][c]);
            }
            __syncthreads();
        }

        // ---- S = Q K^T : four 16x16 score tiles x four K=32 chunks ----
        v8f s[4] = {};
        {
            #pragma unroll
            for (int c = 0; c < 4; ++c) {
                #pragma unroll
                for (int n = 0; n < 4; ++n) {
                    v16bf b = ld_frag(&k_lds[(n * 16 + l16) * KSTR + c * 32 + half * 8]);
                    s[n] = __builtin_amdgcn_wmma_f32_16x16x32_bf16(
                        false, qf[c], false, b, (short)0, s[n], false, false);
                }
            }
            // Pipeline shape: run fragment loads one chunk ahead of the WMMAs.
            __builtin_amdgcn_sched_group_barrier(0x100, 16, 0);  // DS reads
            __builtin_amdgcn_sched_group_barrier(0x008, 4, 0);   // WMMA
            __builtin_amdgcn_sched_group_barrier(0x100, 8, 0);
            __builtin_amdgcn_sched_group_barrier(0x008, 4, 0);
            __builtin_amdgcn_sched_group_barrier(0x100, 8, 0);
            __builtin_amdgcn_sched_group_barrier(0x008, 8, 0);
        }

        // ---- Online softmax in exp2 domain (VALU/DPP only) ----
        #pragma unroll
        for (int r = 0; r < 8; ++r) {
            float a0 = s[0][r] * SCALE_LOG2E;
            float a1 = s[1][r] * SCALE_LOG2E;
            float a2 = s[2][r] * SCALE_LOG2E;
            float a3 = s[3][r] * SCALE_LOG2E;
            float mx = red_max16(fmaxf(fmaxf(a0, a1), fmaxf(a2, a3)));
            float mn = fmaxf(m[r], mx);
            float alpha = __builtin_amdgcn_exp2f(m[r] - mn);
            float p0 = __builtin_amdgcn_exp2f(a0 - mn);
            float p1 = __builtin_amdgcn_exp2f(a1 - mn);
            float p2 = __builtin_amdgcn_exp2f(a2 - mn);
            float p3 = __builtin_amdgcn_exp2f(a3 - mn);
            float sum = red_sum16((p0 + p1) + (p2 + p3));
            l[r] = l[r] * alpha + sum;
            m[r] = mn;
            int row = r + half * 8;
            p_me[row * PSTR +      l16] = f2bf(p0);
            p_me[row * PSTR + 16 + l16] = f2bf(p1);
            p_me[row * PSTR + 32 + l16] = f2bf(p2);
            p_me[row * PSTR + 48 + l16] = f2bf(p3);
            #pragma unroll
            for (int t = 0; t < 8; ++t) o[t][r] *= alpha;
        }

        // In-wave LDS store -> load ordering (per-wave scratch; no barrier).
        asm volatile("s_wait_dscnt 0x0" ::: "memory");

        // ---- P as A fragments: 16x64 -> two K=32 fragments ----
        const __bf16* prow = &p_me[l16 * PSTR];
        v16bf pf0, pf1;
        {
            v8bf a = *(const v8bf*)(prow + half * 8);
            v8bf b = *(const v8bf*)(prow + 16 + half * 8);
            pf0 = __builtin_shufflevector(a, b, 0,1,2,3,4,5,6,7,8,9,10,11,12,13,14,15);
            v8bf c = *(const v8bf*)(prow + 32 + half * 8);
            v8bf d = *(const v8bf*)(prow + 48 + half * 8);
            pf1 = __builtin_shufflevector(c, d, 0,1,2,3,4,5,6,7,8,9,10,11,12,13,14,15);
        }

        // ---- O += P V : 8 output column tiles ----
        {
            #pragma unroll
            for (int t = 0; t < 8; ++t) {
                v16bf bv0 = ld_frag(&vT_lds[(t * 16 + l16) * VSTR + half * 8]);
                v16bf bv1 = ld_frag(&vT_lds[(t * 16 + l16) * VSTR + 32 + half * 8]);
                o[t] = __builtin_amdgcn_wmma_f32_16x16x32_bf16(
                    false, pf0, false, bv0, (short)0, o[t], false, false);
                o[t] = __builtin_amdgcn_wmma_f32_16x16x32_bf16(
                    false, pf1, false, bv1, (short)0, o[t], false, false);
            }
            __builtin_amdgcn_sched_group_barrier(0x100, 12, 1);  // pf + 2 tiles of bv
            #pragma unroll
            for (int g = 0; g < 6; ++g) {
                __builtin_amdgcn_sched_group_barrier(0x008, 2, 1);
                __builtin_amdgcn_sched_group_barrier(0x100, 4, 1);
            }
            __builtin_amdgcn_sched_group_barrier(0x008, 4, 1);
        }
    }

    // ---- In-LDS merge of the two key-split halves, then write out ----
    __syncthreads();   // all LDS tile consumers done; reuse K/V area
    // Merge buffer: [pair 0..3][lane 0..31][81 floats] (81 -> conflict-free)
    float* slot = (float*)smem + (((wave & 3) * 32) + lane) * 81;
    if (grp == 1) {
        #pragma unroll
        for (int t = 0; t < 8; ++t)
            #pragma unroll
            for (int r = 0; r < 8; ++r)
                slot[t * 8 + r] = o[t][r];
        #pragma unroll
        for (int r = 0; r < 8; ++r) { slot[64 + r] = m[r]; slot[72 + r] = l[r]; }
    }
    __syncthreads();
    if (grp == 0) {
        #pragma unroll
        for (int r = 0; r < 8; ++r) {
            float m1 = slot[64 + r];
            float l1 = slot[72 + r];
            float ms = fmaxf(m[r], m1);
            float w0 = __builtin_amdgcn_exp2f(m[r] - ms);
            float w1 = __builtin_amdgcn_exp2f(m1 - ms);
            float inv = 1.0f / (w0 * l[r] + w1 * l1);
            int row = q0 + r + half * 8;
            #pragma unroll
            for (int t = 0; t < 8; ++t)
                O[(size_t)row * DM + t * 16 + l16] =
                    (w0 * o[t][r] + w1 * slot[t * 8 + r]) * inv;
        }
    }
}

extern "C" void kernel_launch(void* const* d_in, const int* in_sizes, int n_in,
                              void* d_out, int out_size, void* d_ws, size_t ws_size,
                              hipStream_t stream) {
    (void)in_sizes; (void)n_in; (void)out_size; (void)d_ws; (void)ws_size;
    const float* Q = (const float*)d_in[0];
    const float* K = (const float*)d_in[1];
    const float* V = (const float*)d_in[2];
    float* Ot = (float*)d_out;
    dim3 grid(SEQ / ROWS_PER_BLOCK);   // 128 blocks
    dim3 block(THREADS);               // 256 threads = 8 waves (2 key groups)
    hipLaunchKernelGGL(SDPAttention_fa_kernel, grid, block, 0, stream, Q, K, V, Ot);
}
